// PointNetSAModule_43714177139080
// MI455X (gfx1250) — matrix operand order repaired and checked by reading, not compiled, not run on hardware
//
#include <hip/hip_runtime.h>
#include <hip/hip_bf16.h>

// ---------------------------------------------------------------------------
// PointNet++ SA module for MI455X (gfx1250, wave32, WMMA).
// GEMMs run on v_wmma_f32_16x16x32_bf16 with f32 accumulation; activations and
// weights are pre-swizzled into WMMA fragment layouts with contiguous k (k = 16*(lane/16)+j). Waves within a
// block share identical B-fragment addresses (differ only in output-channel
// tile) so B loads hit the WGP cache. FPS uses wave32 shfl reductions with
// only 2 block barriers per sequential step.
// ---------------------------------------------------------------------------

typedef __bf16 v16bf __attribute__((ext_vector_type(16)));
typedef float  v8f   __attribute__((ext_vector_type(8)));

static __device__ __forceinline__ unsigned short f2bf(float f) {
    unsigned u = __float_as_uint(f);
    u += 0x7fffu + ((u >> 16) & 1u);          // round-to-nearest-even
    return (unsigned short)(u >> 16);
}
static __device__ __forceinline__ float bf2f(unsigned short h) {
    return __uint_as_float(((unsigned)h) << 16);
}

// ---------------------------------------------------------------------------
__global__ void zero_stats_k(float* stats) {
    int i = blockIdx.x * 256 + threadIdx.x;
    if (i < 3 * 8 * 8 * 2) stats[i] = 0.f;
}

// --------------------------- furthest point sampling -----------------------
// 1 block per batch, 256 threads, 16 points/thread in registers. 1024
// sequential argmax steps. Reduction: wave32 shfl_xor (no barriers) ->
// 8 partials in LDS -> thread 0 folds -> broadcast. 2 barriers per step.
__global__ __launch_bounds__(256) void fps_k(const float* __restrict__ coords,
                                             int* __restrict__ fps_idx) {
    int b = blockIdx.x, t = threadIdx.x;
    int lane = t & 31, wv = t >> 5;
    const float* cx = coords + (size_t)b * 3 * 4096;
    const float* cy = cx + 4096;
    const float* cz = cy + 4096;
    float px[16], py[16], pz[16], md[16];
#pragma unroll
    for (int p = 0; p < 16; p++) {
        int n = p * 256 + t;
        px[p] = cx[n]; py[p] = cy[n]; pz[p] = cz[n];
        md[p] = 3.4e38f;
    }
    __shared__ float wval[8];
    __shared__ int   widx[8];
    __shared__ int   sfar;
    int far = 0;
    for (int s = 0; s < 1024; s++) {
        float sx = cx[far], sy = cy[far], sz = cz[far];
        float bv = -1.f; int bi = 0x7fffffff;
#pragma unroll
        for (int p = 0; p < 16; p++) {
            float dx = px[p] - sx, dy = py[p] - sy, dz = pz[p] - sz;
            float d = dx * dx + dy * dy + dz * dz;
            md[p] = fminf(md[p], d);
            int n = p * 256 + t;
            if (md[p] > bv || (md[p] == bv && n < bi)) { bv = md[p]; bi = n; }
        }
        // wave32 argmax reduce (first-index tie-break)
#pragma unroll
        for (int off = 16; off > 0; off >>= 1) {
            float ov = __shfl_xor(bv, off, 32);
            int   oi = __shfl_xor(bi, off, 32);
            if (ov > bv || (ov == bv && oi < bi)) { bv = ov; bi = oi; }
        }
        if (lane == 0) { wval[wv] = bv; widx[wv] = bi; }
        __syncthreads();
        if (t == 0) {
            float fv = wval[0]; int fi = widx[0];
#pragma unroll
            for (int q = 1; q < 8; q++) {
                float ov = wval[q]; int oi = widx[q];
                if (ov > fv || (ov == fv && oi < fi)) { fv = ov; fi = oi; }
            }
            fps_idx[b * 1024 + s] = far;
            sfar = fi;
        }
        __syncthreads();
        far = sfar;
    }
}

// centers[b][c][s] = coords[b][c][fps_idx[b][s]]  (written straight into d_out)
__global__ void centers_k(const float* __restrict__ coords,
                          const int* __restrict__ fps_idx,
                          float* __restrict__ ctr) {
    int i = blockIdx.x * 256 + threadIdx.x;           // B*3*S = 24576
    if (i >= 24576) return;
    int s = i & 1023;
    int c = (i >> 10) % 3;
    int b = i / 3072;
    ctr[i] = coords[((size_t)b * 3 + c) * 4096 + fps_idx[b * 1024 + s]];
}

// --------------------------- ball query (1 wave32 / center) ----------------
__global__ __launch_bounds__(256) void ballq_k(const float* __restrict__ coords,
                                               const float* __restrict__ centers,
                                               int* __restrict__ nidx) {
    int wid   = (blockIdx.x * 256 + threadIdx.x) >> 5;  // 8192 centers
    int lane  = threadIdx.x & 31;
    int lwave = threadIdx.x >> 5;
    __shared__ int buf[8][32];
    int b = wid >> 10, s = wid & 1023;
    const float* cx = coords + (size_t)b * 3 * 4096;
    float ccx = centers[(b * 3 + 0) * 1024 + s];
    float ccy = centers[(b * 3 + 1) * 1024 + s];
    float ccz = centers[(b * 3 + 2) * 1024 + s];
    int count = 0;
    for (int n0 = 0; n0 < 4096 && count < 32; n0 += 32) {
        int n = n0 + lane;
        float dx = cx[n] - ccx, dy = cx[4096 + n] - ccy, dz = cx[8192 + n] - ccz;
        bool valid = (dx * dx + dy * dy + dz * dz) < 0.0225f;   // R^2
        unsigned mask = (unsigned)__ballot(valid);
        int slot = count + __popc(mask & ((1u << lane) - 1u));
        if (valid && slot < 32) buf[lwave][slot] = n;
        count += __popc(mask);
        if (count > 32) count = 32;
    }
    __syncthreads();
    int v;
    if (count == 0)        v = 0;
    else if (lane < count) v = buf[lwave][lane];
    else                   v = buf[lwave][0];
    nidx[(size_t)wid * 32 + lane] = v;
}

// -------------------- weight swizzle into WMMA A-fragment layout -----------
// A 16x32 bf16 layout: row = lane&15; ci = kb*32 + v*2+h + (lane>=16?16:0)
__global__ void swz_w_k(const float* __restrict__ w, unsigned short* __restrict__ wsz,
                        int CI, int CO, int KB) {
    int total = (CO / 16) * KB * 512;
    int i = blockIdx.x * 256 + threadIdx.x;
    if (i >= total) return;
    int j = i & 15, lane = (i >> 4) & 31;
    int kb = (i >> 9) % KB, cot = (i >> 9) / KB;
    int v = j >> 1, h = j & 1;
    int ci = kb * 32 + v * 2 + h + ((lane & 16) ? 16 : 0);
    int co = cot * 16 + (lane & 15);
    float val = (ci < CI) ? w[(size_t)co * CI + ci] : 0.f;
    wsz[i] = f2bf(val);
}

// ------------- gather (ncoords|features) into WMMA B-fragment layout -------
// B 32x16 bf16 layout: col = lane&15; k = v*2+h + (lane>=16?16:0)
// X layout: [b][kb(KB=5)][mt(2048)][lane(32)][16 halfs]
__global__ void build_x0_k(const float* __restrict__ coords,
                           const float* __restrict__ features,
                           const float* __restrict__ ctr,
                           const int* __restrict__ nidx,
                           unsigned short* __restrict__ xs) {
    size_t i = (size_t)blockIdx.x * 256 + threadIdx.x;  // 8*5*2048*512
    int j = (int)(i & 15), lane = (int)((i >> 4) & 31), mt = (int)((i >> 9) & 2047);
    size_t r = i >> 20;
    int kb = (int)(r % 5), b = (int)(r / 5);
    int v = j >> 1, h = j & 1;
    int ci = kb * 32 + v * 2 + h + ((lane & 16) ? 16 : 0);
    int m = mt * 16 + (lane & 15);
    int s = m >> 5, k = m & 31;
    float val = 0.f;
    if (ci < 131) {
        int n = nidx[((size_t)(b * 1024 + s)) * 32 + k];
        if (ci < 3) val = coords[((size_t)b * 3 + ci) * 4096 + n] - ctr[(b * 3 + ci) * 1024 + s];
        else        val = features[((size_t)b * 128 + (ci - 3)) * 4096 + n];
    }
    xs[i] = f2bf(val);
}

// --------------------------- WMMA GEMM + bias + stats ----------------------
// Wave mapping: the 8 waves of a block share the same m-chunk and differ in
// the output-channel tile, so all 8 issue identical B-fragment loads (WGP$
// hits) while each keeps its own A fragments resident in VGPRs. The flattened
// (tile x kb) stream is software-pipelined depth-2 so B loads stay in flight
// across WMMAs. GroupNorm sum/sumsq accumulate via wave reduce + f32 atomics.
template <int KB, int CO>
__global__ __launch_bounds__(256) void gemm_k(const unsigned short* __restrict__ xs,
                                              const unsigned short* __restrict__ wsz,
                                              const float* __restrict__ bias,
                                              unsigned short* __restrict__ y,
                                              float* __restrict__ stats) {
    int w    = (blockIdx.x << 3) + (threadIdx.x >> 5);
    int lane = threadIdx.x & 31;
    int cotl   = w & 7;                 // differs across waves in a block
    int rest   = w >> 3;
    int mchunk = rest & 255;            // shared by all waves in a block
    int rest2  = rest >> 8;
    int coth   = rest2 % (CO / 128);
    int b      = rest2 / (CO / 128);
    int cot    = coth * 8 + cotl;

    v16bf a[KB];
#pragma unroll
    for (int kb = 0; kb < KB; kb++)
        a[kb] = *reinterpret_cast<const v16bf*>(wsz + ((size_t)(cot * KB + kb) * 512 + lane * 16));

    int baseco = cot * 16 + ((lane & 16) ? 8 : 0);
    float bi[8];
#pragma unroll
    for (int v = 0; v < 8; v++) bi[v] = bias[baseco + v];

    const unsigned short* xb = xs + (size_t)b * KB * 2048 * 512 + lane * 16;
    const int mt0 = mchunk * 8;
    // fragment address for flattened step i = t*KB + kb
    auto baddr = [&](int i) -> const v16bf* {
        int t = i / KB, kb = i % KB;
        return reinterpret_cast<const v16bf*>(xb + ((size_t)kb * 2048 + (mt0 + t)) * 512);
    };

    constexpr int NS = 8 * KB;
    v16bf bA = *baddr(0);
    v16bf bB = *baddr(1);
    v8f c = {0.f, 0.f, 0.f, 0.f, 0.f, 0.f, 0.f, 0.f};
    float s1 = 0.f, s2 = 0.f;
#pragma unroll
    for (int i = 0; i < NS; i++) {
        v16bf bN;
        if (i + 2 < NS) bN = *baddr(i + 2);
        else            bN = bA;
        c = __builtin_amdgcn_wmma_f32_16x16x32_bf16(false, a[i % KB], false, bA,
                                                    (short)0, c, false, false);
        bA = bB; bB = bN;
        if (i % KB == KB - 1) {                       // tile finished
            int t = i / KB;
            int m = (mt0 + t) * 16 + (lane & 15);
#pragma unroll
            for (int v = 0; v < 8; v++) {
                float yv = c[v] + bi[v];
                s1 += yv; s2 += yv * yv;
                y[((size_t)b * CO + baseco + v) * 32768 + m] = f2bf(yv);
            }
            c = (v8f){0.f, 0.f, 0.f, 0.f, 0.f, 0.f, 0.f, 0.f};
        }
    }
#pragma unroll
    for (int off = 16; off > 0; off >>= 1) {
        s1 += __shfl_xor(s1, off, 32);
        s2 += __shfl_xor(s2, off, 32);
    }
    if (lane == 0) {
        int g = (cot * 16) / (CO / 8);
        unsafeAtomicAdd(&stats[(b * 8 + g) * 2 + 0], s1);
        unsafeAtomicAdd(&stats[(b * 8 + g) * 2 + 1], s2);
    }
}

// ------------- GroupNorm + SiLU, re-swizzled to next B-fragment X ----------
__global__ void norm12_k(const unsigned short* __restrict__ y,
                         unsigned short* __restrict__ xs,
                         const float* __restrict__ gamma,
                         const float* __restrict__ beta,
                         const float* __restrict__ stats) {
    size_t i = (size_t)blockIdx.x * 256 + threadIdx.x;   // 8*4*2048*512
    int j = (int)(i & 15), lane = (int)((i >> 4) & 31), mt = (int)((i >> 9) & 2047);
    size_t r = i >> 20;
    int kb = (int)(r & 3), b = (int)(r >> 2);
    int v = j >> 1, h = j & 1;
    int c = kb * 32 + v * 2 + h + ((lane & 16) ? 16 : 0);   // CO = 128
    int m = mt * 16 + (lane & 15);
    int g = c >> 4;                                         // 16 ch / group
    float sum = stats[(b * 8 + g) * 2], sq = stats[(b * 8 + g) * 2 + 1];
    const float cnt = 16.f * 32768.f;
    float mu = sum / cnt;
    float var = sq / cnt - mu * mu;
    float rs = rsqrtf(var + 1e-5f);
    float yv = bf2f(y[((size_t)b * 128 + c) * 32768 + m]);
    float xn = (yv - mu) * rs * gamma[c] + beta[c];
    float sl = xn / (1.f + __expf(-xn));
    xs[i] = f2bf(sl);
}

// -------- layer-3 GroupNorm + SiLU + max over K, writes final output -------
__global__ void final3_k(const unsigned short* __restrict__ y,
                         const float* __restrict__ gamma,
                         const float* __restrict__ beta,
                         const float* __restrict__ stats,
                         float* __restrict__ out) {
    int i = blockIdx.x * 256 + threadIdx.x;   // B*256*S = 2,097,152
    int s = i & 1023, c = (i >> 10) & 255, b = i >> 18;
    int g = c >> 5;                            // 32 ch / group (CO=256)
    float sum = stats[(b * 8 + g) * 2], sq = stats[(b * 8 + g) * 2 + 1];
    const float cnt = 32.f * 32768.f;
    float mu = sum / cnt;
    float var = sq / cnt - mu * mu;
    float rs = rsqrtf(var + 1e-5f);
    float ga = gamma[c] * rs;
    float be = beta[c] - mu * ga;
    float mx = -3.4e38f;
    const unsigned short* yp = y + ((size_t)b * 256 + c) * 32768 + s * 32;
#pragma unroll 8
    for (int k = 0; k < 32; k++) {
        float xn = bf2f(yp[k]) * ga + be;
        float sl = xn / (1.f + __expf(-xn));
        mx = fmaxf(mx, sl);
    }
    out[i] = mx;
}

// --------------------------- passthrough outputs ---------------------------
__global__ void copy_te_k(const float* __restrict__ te, float* __restrict__ o) {
    int i = blockIdx.x * 256 + threadIdx.x;   // B*64*S = 524288
    int s = i & 1023, bc = i >> 10;
    o[i] = te[(size_t)bc * 4096 + s];
}
__global__ void copy_sty_k(const float* __restrict__ st, float* __restrict__ o) {
    int i = blockIdx.x * 256 + threadIdx.x;
    if (i < 512) o[i] = st[i];
}

// ---------------------------------------------------------------------------
extern "C" void kernel_launch(void* const* d_in, const int* in_sizes, int n_in,
                              void* d_out, int out_size, void* d_ws, size_t ws_size,
                              hipStream_t stream) {
    const float* features = (const float*)d_in[0];
    const float* coords   = (const float*)d_in[1];
    const float* time_emb = (const float*)d_in[2];
    const float* style    = (const float*)d_in[3];
    const float* w1 = (const float*)d_in[4];  const float* b1  = (const float*)d_in[5];
    const float* g1 = (const float*)d_in[6];  const float* be1 = (const float*)d_in[7];
    const float* w2 = (const float*)d_in[8];  const float* b2  = (const float*)d_in[9];
    const float* g2 = (const float*)d_in[10]; const float* be2 = (const float*)d_in[11];
    const float* w3 = (const float*)d_in[12]; const float* b3  = (const float*)d_in[13];
    const float* g3 = (const float*)d_in[14]; const float* be3 = (const float*)d_in[15];

    float* out  = (float*)d_out;          // [8,256,1024]
    float* ctr  = out + 2097152;          // [8,3,1024]
    float* ote  = ctr + 24576;            // [8,64,1024]
    float* osty = ote + 524288;           // [8,64]

    char* ws = (char*)d_ws;
    size_t off = 0;
    auto alloc = [&](size_t bytes) -> void* {
        void* p = ws + off;
        off = (off + bytes + 255) & ~(size_t)255;
        return p;
    };
    int*   fpsidx = (int*)alloc(8ull * 1024 * 4);
    int*   nidx   = (int*)alloc(8ull * 1024 * 32 * 4);
    float* stats  = (float*)alloc(3ull * 8 * 8 * 2 * 4);
    unsigned short* ws1  = (unsigned short*)alloc(8ull * 5 * 512 * 2);
    unsigned short* ws2  = (unsigned short*)alloc(8ull * 4 * 512 * 2);
    unsigned short* ws3  = (unsigned short*)alloc(16ull * 4 * 512 * 2);
    unsigned short* xbuf = (unsigned short*)alloc(8ull * 5 * 2048 * 512 * 2);   // 83.9 MB
    unsigned short* ybuf = (unsigned short*)alloc(8ull * 256 * 32768 * 2);      // 134 MB

    zero_stats_k<<<2, 256, 0, stream>>>(stats);
    fps_k<<<8, 256, 0, stream>>>(coords, fpsidx);
    centers_k<<<96, 256, 0, stream>>>(coords, fpsidx, ctr);
    ballq_k<<<1024, 256, 0, stream>>>(coords, ctr, nidx);
    swz_w_k<<<80, 256, 0, stream>>>(w1, ws1, 131, 128, 5);
    swz_w_k<<<64, 256, 0, stream>>>(w2, ws2, 128, 128, 4);
    swz_w_k<<<128, 256, 0, stream>>>(w3, ws3, 128, 256, 4);
    build_x0_k<<<163840, 256, 0, stream>>>(coords, features, ctr, nidx, xbuf);

    gemm_k<5, 128><<<2048, 256, 0, stream>>>(xbuf, ws1, b1, ybuf, stats + 0);
    norm12_k<<<131072, 256, 0, stream>>>(ybuf, xbuf, g1, be1, stats + 0);
    gemm_k<4, 128><<<2048, 256, 0, stream>>>(xbuf, ws2, b2, ybuf, stats + 128);
    norm12_k<<<131072, 256, 0, stream>>>(ybuf, xbuf, g2, be2, stats + 128);
    gemm_k<4, 256><<<4096, 256, 0, stream>>>(xbuf, ws3, b3, ybuf, stats + 256);
    final3_k<<<8192, 256, 0, stream>>>(ybuf, g3, be3, stats + 256, out);

    copy_te_k<<<2048, 256, 0, stream>>>(time_emb, ote);
    copy_sty_k<<<2, 256, 0, stream>>>(style, osty);
}
